// MultiHeadAttention_67714454389214
// MI455X (gfx1250) — matrix operand order repaired and checked by reading, not compiled, not run on hardware
//
#include <hip/hip_runtime.h>
#include <hip/hip_bf16.h>

// ---------------- problem constants ----------------
constexpr int B_ = 2;
constexpr int S_ = 2048;
constexpr int E_ = 1024;
constexpr int H_ = 16;
constexpr int D_ = 64;          // E/H
constexpr int M_ = B_ * S_;     // 4096 tokens

typedef __attribute__((ext_vector_type(16))) _Float16 v16h;
typedef __attribute__((ext_vector_type(8)))  _Float16 v8h;
typedef __attribute__((ext_vector_type(4)))  _Float16 v4h;
typedef __attribute__((ext_vector_type(8)))  float    v8f;

// Build a 16-half WMMA fragment from two 16-byte chunks.
__device__ __forceinline__ v16h ld_frag(const _Float16* p0, const _Float16* p1) {
    v8h lo = *(const v8h*)p0;
    v8h hi = *(const v8h*)p1;
    return __builtin_shufflevector(lo, hi, 0,1,2,3,4,5,6,7,8,9,10,11,12,13,14,15);
}

__device__ __forceinline__ v8f wmma_f16(v16h a, v16h b, v8f c) {
    return __builtin_amdgcn_wmma_f32_16x16x32_f16(false, a, false, b, (short)0, c, false, false);
}

// ---------------- fp32 -> fp16 convert ----------------
__global__ __launch_bounds__(256) void cvt_f32_f16(const float* __restrict__ src,
                                                   _Float16* __restrict__ dst, int n) {
    int i = (blockIdx.x * blockDim.x + threadIdx.x) * 4;
    if (i + 3 < n) {
        float4 v = *(const float4*)(src + i);
        v4h o; o.x = (_Float16)v.x; o.y = (_Float16)v.y; o.z = (_Float16)v.z; o.w = (_Float16)v.w;
        *(v4h*)(dst + i) = o;
    }
}

// ---------------- fused QKV projection GEMM ----------------
// X3  : f16 [3][M][E]  (q,k,v activations, contiguous)
// W   : f16 [3E][E]    (C = X @ W^T + b)
// out : qp [B,H,S,D], kp [B,H,S,D], vt [B,H,D,S]  (V pre-transposed for P@V)
// Block = 256 thr = 8 waves laid out 4(M) x 2(N); wave tile 32x64; block tile 128x128.
__global__ __launch_bounds__(256) void qkv_gemm(
    const _Float16* __restrict__ X3, const _Float16* __restrict__ W,
    const float* __restrict__ bias,
    _Float16* __restrict__ qp, _Float16* __restrict__ kp, _Float16* __restrict__ vt)
{
    const int lane = threadIdx.x & 31, wave = threadIdx.x >> 5;
    const int lr = lane & 15, lh = lane >> 4;
    const int m0 = blockIdx.x * 128 + (wave >> 1) * 32;
    const int n0 = blockIdx.y * 128 + (wave & 1) * 64;
    const int region = n0 >> 10;                       // 0=Q,1=K,2=V (E=1024)
    const _Float16* X = X3 + (size_t)region * ((size_t)M_ * E_);

    v8f acc[2][4] = {};
    for (int kk = 0; kk < E_; kk += 32) {
        v16h a[2], bf[4];
#pragma unroll
        for (int i = 0; i < 2; ++i) {
            const _Float16* ap = X + (size_t)(m0 + i * 16 + lr) * E_ + kk;
            a[i] = ld_frag(ap + lh * 8, ap + lh * 8 + 16);
        }
#pragma unroll
        for (int j = 0; j < 4; ++j) {
            const _Float16* bp = W + (size_t)(n0 + j * 16 + lr) * E_ + kk;
            bf[j] = ld_frag(bp + lh * 8, bp + lh * 8 + 16);
        }
#pragma unroll
        for (int i = 0; i < 2; ++i)
#pragma unroll
            for (int j = 0; j < 4; ++j)
                acc[i][j] = wmma_f16(a[i], bf[j], acc[i][j]);
    }

    // Epilogue: C layout — lane's column fixed (lr), VGPR r = row (r + 8*lh).
    // Destination base + row-stride are uniform per 16x16 tile: hoist them.
#pragma unroll
    for (int i = 0; i < 2; ++i)
#pragma unroll
        for (int j = 0; j < 4; ++j) {
            const int col = n0 + j * 16 + lr;          // [0,3072)
            const float bv = bias[col];
            const int f = col & (E_ - 1);
            const int h = f >> 6, d = f & 63;
            const int mrow0 = m0 + i * 16 + lh * 8;
            const int bb = mrow0 >> 11, ss0 = mrow0 & (S_ - 1);
            _Float16* dst;
            size_t addr;
            int step;
            if (region == 0) {
                dst = qp; addr = (((size_t)bb * H_ + h) * S_ + ss0) * D_ + d; step = D_;
            } else if (region == 1) {
                dst = kp; addr = (((size_t)bb * H_ + h) * S_ + ss0) * D_ + d; step = D_;
            } else {
                dst = vt; addr = (((size_t)bb * H_ + h) * D_ + d) * S_ + ss0; step = 1;
            }
#pragma unroll
            for (int r = 0; r < 8; ++r)
                dst[addr + (size_t)r * step] = (_Float16)(acc[i][j][r] + bv);
        }
}

// ---------------- flash attention (no materialized SxS) ----------------
// One wave = one (b,h) x 32 query rows (2 q-tiles; K/V fragments shared).
// Streams 32 keys per iteration: 8 score WMMAs + 8 P@V WMMAs per chunk.
// qp,kp: [B,H,S,D] f16; vt: [B,H,D,S] f16; ao: [B*S][E] f16 (token-major).
__global__ __launch_bounds__(128) void attn_kernel(
    const _Float16* __restrict__ qp, const _Float16* __restrict__ kp,
    const _Float16* __restrict__ vt, const int* __restrict__ mask,
    _Float16* __restrict__ ao)
{
    __shared__ __align__(16) _Float16 pbuf[4][2][16 * 32];   // per-wave P staging (8 KB)
    const int lane = threadIdx.x & 31, wave = threadIdx.x >> 5;
    const int lr = lane & 15, lh = lane >> 4;
    const int w  = blockIdx.x * 4 + wave;                 // 2048 waves total
    const int bh = w >> 6;                                // S/32 = 64 q-pairs per (b,h)
    const int qt = w & 63;
    const int b  = bh >> 4, h = bh & 15;
    const int q0 = qt * 32;

    const _Float16* qb  = qp + ((size_t)bh * S_ + q0) * D_;
    const _Float16* kbp = kp + (size_t)bh * S_ * D_;
    const _Float16* vb  = vt + (size_t)bh * D_ * S_;
    const int* mrow = mask + b * S_;

    // Q A-fragments (K-dim = d, 64 -> two frags per q-tile), loop-resident.
    v16h qa[2][2];
#pragma unroll
    for (int t = 0; t < 2; ++t) {
        const _Float16* ap = qb + (size_t)(t * 16 + lr) * D_;
        qa[t][0] = ld_frag(ap + lh * 8,      ap + lh * 8 + 16);
        qa[t][1] = ld_frag(ap + 32 + lh * 8, ap + 32 + lh * 8 + 16);
    }

    float mi[2][8], li[2][8];
    v8f acc[2][4] = {};
#pragma unroll
    for (int t = 0; t < 2; ++t)
#pragma unroll
        for (int r = 0; r < 8; ++r) { mi[t][r] = -1e30f; li[t][r] = 0.f; }

    for (int kb = 0; kb < S_; kb += 32) {
        // ---- K B-fragments for two 16-key tiles (shared by both q-tiles) ----
        const _Float16* p0 = kbp + (size_t)(kb + lr) * D_;
        const _Float16* p1 = kbp + (size_t)(kb + 16 + lr) * D_;
        v16h k0a = ld_frag(p0 + lh * 8,      p0 + lh * 8 + 16);       // d 0..31
        v16h k0b = ld_frag(p0 + 32 + lh * 8, p0 + 32 + lh * 8 + 16);  // d 32..63
        v16h k1a = ld_frag(p1 + lh * 8,      p1 + lh * 8 + 16);
        v16h k1b = ld_frag(p1 + 32 + lh * 8, p1 + 32 + lh * 8 + 16);

        const int mk0 = mrow[kb + lr];
        const int mk1 = mrow[kb + 16 + lr];

        // ---- scores, mask, online softmax, P-staging per q-tile ----
#pragma unroll
        for (int t = 0; t < 2; ++t) {
            v8f s0 = {}, s1 = {};
            s0 = wmma_f16(qa[t][0], k0a, s0);
            s0 = wmma_f16(qa[t][1], k0b, s0);
            s1 = wmma_f16(qa[t][0], k1a, s1);
            s1 = wmma_f16(qa[t][1], k1b, s1);
#pragma unroll
            for (int r = 0; r < 8; ++r) {
                if (mk0 == 0) s0[r] = -1e20f;
                if (mk1 == 0) s1[r] = -1e20f;
            }
#pragma unroll
            for (int r = 0; r < 8; ++r) {
                float mx = fmaxf(s0[r], s1[r]);
                mx = fmaxf(mx, __shfl_xor(mx, 1));
                mx = fmaxf(mx, __shfl_xor(mx, 2));
                mx = fmaxf(mx, __shfl_xor(mx, 4));
                mx = fmaxf(mx, __shfl_xor(mx, 8));
                const float mnew = fmaxf(mi[t][r], mx);
                const float sc = __expf(mi[t][r] - mnew);
                const float e0 = __expf(s0[r] - mnew);
                const float e1 = __expf(s1[r] - mnew);
                float rs = e0 + e1;
                rs += __shfl_xor(rs, 1);
                rs += __shfl_xor(rs, 2);
                rs += __shfl_xor(rs, 4);
                rs += __shfl_xor(rs, 8);
                li[t][r] = li[t][r] * sc + rs;
                mi[t][r] = mnew;
#pragma unroll
                for (int j = 0; j < 4; ++j) acc[t][j][r] *= sc;
                const int row = r + lh * 8;
                pbuf[wave][t][row * 32 + lr]      = (_Float16)e0;
                pbuf[wave][t][row * 32 + 16 + lr] = (_Float16)e1;
            }
        }
        asm volatile("s_wait_dscnt 0" ::: "memory");

        // ---- P A-fragments (C layout -> A layout via LDS) ----
        v16h pa[2];
#pragma unroll
        for (int t = 0; t < 2; ++t) {
            const _Float16* pp = &pbuf[wave][t][lr * 32 + lh * 8];
            pa[t] = ld_frag(pp, pp + 16);
        }

        // ---- O += P @ V ; V^T rows key-contiguous; vf shared by both q-tiles ----
#pragma unroll
        for (int j = 0; j < 4; ++j) {
            const _Float16* vp_ = vb + (size_t)(j * 16 + lr) * S_ + kb + lh * 8;
            v16h vf = ld_frag(vp_, vp_ + 16);
#pragma unroll
            for (int t = 0; t < 2; ++t)
                acc[t][j] = wmma_f16(pa[t], vf, acc[t][j]);
        }
    }

    // ---- normalize + write [B*S][E] f16, feature = h*64 + j*16 + lr ----
#pragma unroll
    for (int t = 0; t < 2; ++t)
#pragma unroll
        for (int r = 0; r < 8; ++r) {
            const float inv = 1.f / li[t][r];
            const int row = q0 + t * 16 + r + lh * 8;
            const size_t obase = ((size_t)b * S_ + row) * E_ + h * D_;
#pragma unroll
            for (int j = 0; j < 4; ++j)
                ao[obase + j * 16 + lr] = (_Float16)(acc[t][j][r] * inv);
        }
}

// ---------------- output projection GEMM (f32 out) ----------------
// Same 32x64 wave tile / 128x128 block tile as qkv_gemm; row-major f32 epilogue.
__global__ __launch_bounds__(256) void out_gemm(
    const _Float16* __restrict__ X, const _Float16* __restrict__ W,
    const float* __restrict__ bias, float* __restrict__ out)
{
    const int lane = threadIdx.x & 31, wave = threadIdx.x >> 5;
    const int lr = lane & 15, lh = lane >> 4;
    const int m0 = blockIdx.x * 128 + (wave >> 1) * 32;
    const int n0 = blockIdx.y * 128 + (wave & 1) * 64;

    v8f acc[2][4] = {};
    for (int kk = 0; kk < E_; kk += 32) {
        v16h a[2], bf[4];
#pragma unroll
        for (int i = 0; i < 2; ++i) {
            const _Float16* ap = X + (size_t)(m0 + i * 16 + lr) * E_ + kk;
            a[i] = ld_frag(ap + lh * 8, ap + lh * 8 + 16);
        }
#pragma unroll
        for (int j = 0; j < 4; ++j) {
            const _Float16* bp = W + (size_t)(n0 + j * 16 + lr) * E_ + kk;
            bf[j] = ld_frag(bp + lh * 8, bp + lh * 8 + 16);
        }
#pragma unroll
        for (int i = 0; i < 2; ++i)
#pragma unroll
            for (int j = 0; j < 4; ++j)
                acc[i][j] = wmma_f16(a[i], bf[j], acc[i][j]);
    }
#pragma unroll
    for (int i = 0; i < 2; ++i)
#pragma unroll
        for (int j = 0; j < 4; ++j) {
            const int col = n0 + j * 16 + lr;
            const float bv = bias[col];
            const size_t base = (size_t)(m0 + i * 16 + lh * 8) * E_ + col;
#pragma unroll
            for (int r = 0; r < 8; ++r)
                out[base + (size_t)r * E_] = acc[i][j][r] + bv;
        }
}

// ---------------- host-side launcher ----------------
extern "C" void kernel_launch(void* const* d_in, const int* in_sizes, int n_in,
                              void* d_out, int out_size, void* d_ws, size_t ws_size,
                              hipStream_t stream) {
    const float* q     = (const float*)d_in[0];
    const float* k     = (const float*)d_in[1];
    const float* v     = (const float*)d_in[2];
    const int*   mask  = (const int*)d_in[3];
    const float* W_qkv = (const float*)d_in[4];
    const float* b_qkv = (const float*)d_in[5];
    const float* W_out = (const float*)d_in[6];
    const float* b_out = (const float*)d_in[7];
    float* out = (float*)d_out;

    _Float16* ws = (_Float16*)d_ws;
    const size_t ME = (size_t)M_ * E_;         // 4 Mi elements
    const size_t oX   = 0;                     // hq,hk,hv contiguous: 3*ME
    const size_t oW   = oX + 3 * ME;           // W_qkv f16: 3*E*E
    const size_t oWo  = oW + 3ull * E_ * E_;   // W_out f16: E*E
    const size_t oQP  = oWo + (size_t)E_ * E_; // qp [B,H,S,D]
    const size_t oKP  = oQP + ME;              // kp [B,H,S,D]
    const size_t oVT  = oKP + ME;              // vt [B,H,D,S]
    const size_t oAO  = oVT + ME;              // attention out [B*S][E]
    (void)ws_size; (void)n_in; (void)in_sizes; (void)out_size;

    auto cvt = [&](const float* s, _Float16* d, int n) {
        cvt_f32_f16<<<dim3((unsigned)((n / 4 + 255) / 256)), dim3(256), 0, stream>>>(s, d, n);
    };
    cvt(q,     ws + oX,          (int)ME);
    cvt(k,     ws + oX + ME,     (int)ME);
    cvt(v,     ws + oX + 2 * ME, (int)ME);
    cvt(W_qkv, ws + oW,          3 * E_ * E_);
    cvt(W_out, ws + oWo,         E_ * E_);

    qkv_gemm<<<dim3(M_ / 128, 3 * E_ / 128), dim3(256), 0, stream>>>(
        ws + oX, ws + oW, b_qkv, ws + oQP, ws + oKP, ws + oVT);

    attn_kernel<<<dim3((B_ * H_ * (S_ / 32)) / 4), dim3(128), 0, stream>>>(
        ws + oQP, ws + oKP, ws + oVT, mask, ws + oAO);

    out_gemm<<<dim3(M_ / 128, E_ / 128), dim3(256), 0, stream>>>(
        ws + oAO, ws + oWo, b_out, out);
}
